// CapsuleLayer_69002944578095
// MI455X (gfx1250) — matrix operand order repaired and checked by reading, not compiled, not run on hardware
//
#include <hip/hip_runtime.h>
#include <hip/hip_bf16.h>

// ---------------------------------------------------------------------------
// CapsuleLayer for MI455X (gfx1250, wave32, WMMA)
//   inputs_hat = x[16384,1024] @ kernel[1024,512]   (bf16x3-compensated WMMA)
//   3 rounds of dynamic routing on [32,16] capsules (wave32 shuffle kernel)
// ---------------------------------------------------------------------------

typedef __attribute__((ext_vector_type(16))) __bf16 v16bf;
typedef __attribute__((ext_vector_type(8)))  float  v8f;
typedef __attribute__((ext_vector_type(4)))  float  v4f;

#define M_TOT  16384
#define K_TOT  1024
#define N_TOT  512          // NUM_CAPSULE * DIM_CAPSULE
#define NCAP   32
#define DCAP   16

// ---------------------------------------------------------------------------
// Kernel 0: convert B (f32 row-major [K,N]) into bf16 hi/lo fragment layout.
// Fragment layout for V_WMMA_F32_16X16X32_BF16 B operand (32x16, K x N):
//   lane L (0..31) holds column N = L&15 of the tile; its 8 VGPRs = 16 bf16 = 16
//   K values 16*(e>>3)+8*(L>>4)+(e&7) of that column.  We store each fragment as 16 contiguous
//   bf16 per lane so the GEMM reads it with plain 128-bit loads.
//   Bsw[((kt*32 + nt)*32 + lane)*16 + e] = bf16split(B[kt*32+16*(e>>3)+8*(lane>>4)+(e&7)][nt*16+(lane&15)])
// ---------------------------------------------------------------------------
__global__ __launch_bounds__(256) void capsule_swizzle_b(
    const float* __restrict__ B, __bf16* __restrict__ Bhi, __bf16* __restrict__ Blo) {
  int t    = blockIdx.x * 256 + threadIdx.x;   // 32768 threads total
  int lane = t & 31;
  int nt   = (t >> 5) & 31;                    // N tile (0..31)
  int kt   = t >> 10;                          // K tile (0..31)
  const float* src = B + (size_t)(kt * 32) * N_TOT + nt * 16 + (lane & 15);
  size_t fo = (((size_t)kt * 32 + nt) * 32 + lane) * 16;
#pragma unroll
  for (int j = 0; j < 16; ++j) {
    float a  = src[(size_t)(((j & 8) << 1) + ((lane >> 4) << 3) + (j & 7)) * N_TOT];
    __bf16 h = (__bf16)a;                      // RNE high part
    Bhi[fo + j] = h;
    Blo[fo + j] = (__bf16)(a - (float)h);      // residual
  }
}

// ---------------------------------------------------------------------------
// Kernel 1: GEMM, bf16x3 compensated, f32 accumulation in WMMA.
//   grid = (N_TOT/256, M_TOT/128), block = 256 threads = 8 waves
//   wave tile = 32(M) x 128(N) = 2 x 8 tiles of 16x16; K stepped by 32.
// A-fragment ISA layout (16-bit A 16x32): lane = {kh = L>>4, m = L&15};
//   elements 0..7  = A[m][kb + kh*8 + 0..7]
//   elements 8..15 = A[m][kb + 16 + kh*8 + 0..7]
// ---------------------------------------------------------------------------
__global__ __launch_bounds__(256) void capsule_gemm(
    const float*  __restrict__ A,
    const __bf16* __restrict__ Bhi,
    const __bf16* __restrict__ Blo,
    float*        __restrict__ hat) {
  const int lane = threadIdx.x & 31;
  const int wave = threadIdx.x >> 5;
  const int wm   = wave & 3;          // 4 waves along M
  const int wn   = wave >> 2;         // 2 waves along N
  const int m0   = blockIdx.y * 128 + wm * 32;
  const int n0   = blockIdx.x * 256 + wn * 128;
  const int kh   = lane >> 4;         // K-half select for A fragment
  const int ml   = lane & 15;         // M row within 16x16 tile

  v8f acc[2][8] = {};

  for (int kb = 0; kb < K_TOT; kb += 32) {
    // ---- load + split A fragments for the two M sub-tiles ----
    v16bf ahi[2], alo[2];
#pragma unroll
    for (int mi = 0; mi < 2; ++mi) {
      const float* ap = A + (size_t)(m0 + mi * 16 + ml) * K_TOT + kb + kh * 8;
      v4f r0 = *(const v4f*)(ap +  0);
      v4f r1 = *(const v4f*)(ap +  4);
      v4f r2 = *(const v4f*)(ap + 16);
      v4f r3 = *(const v4f*)(ap + 20);
      float af[16];
#pragma unroll
      for (int j = 0; j < 4; ++j) {
        af[j] = r0[j]; af[4 + j] = r1[j]; af[8 + j] = r2[j]; af[12 + j] = r3[j];
      }
#pragma unroll
      for (int j = 0; j < 16; ++j) {
        __bf16 h = (__bf16)af[j];
        ahi[mi][j] = h;
        alo[mi][j] = (__bf16)(af[j] - (float)h);
      }
    }

    const int kt = kb >> 5;
    // ---- 8 N tiles x 2 M tiles x 3 compensated WMMAs = 48 wmma / K-step ----
#pragma unroll
    for (int ni = 0; ni < 8; ++ni) {
      const int nt = (n0 >> 4) + ni;
      const size_t fo = (((size_t)kt * 32 + nt) * 32 + lane) * 16;
      const v16bf bhi = *(const v16bf*)(Bhi + fo);
      const v16bf blo = *(const v16bf*)(Blo + fo);
#pragma unroll
      for (int mi = 0; mi < 2; ++mi) {
        acc[mi][ni] = __builtin_amdgcn_wmma_f32_16x16x32_bf16(
            false, alo[mi], false, bhi, (short)0, acc[mi][ni], false, false);
        acc[mi][ni] = __builtin_amdgcn_wmma_f32_16x16x32_bf16(
            false, ahi[mi], false, blo, (short)0, acc[mi][ni], false, false);
        acc[mi][ni] = __builtin_amdgcn_wmma_f32_16x16x32_bf16(
            false, ahi[mi], false, bhi, (short)0, acc[mi][ni], false, false);
      }
    }
  }

  // ---- epilogue: C/D layout: reg r, lane -> M = r + 8*kh, N = ml ----
#pragma unroll
  for (int mi = 0; mi < 2; ++mi)
#pragma unroll
    for (int ni = 0; ni < 8; ++ni)
#pragma unroll
      for (int r = 0; r < 8; ++r) {
        int mrow = m0 + mi * 16 + (kh << 3) + r;
        int ncol = n0 + ni * 16 + ml;
        hat[(size_t)mrow * N_TOT + ncol] = acc[mi][ni][r];
      }
}

// ---------------------------------------------------------------------------
// Kernel 2: dynamic routing. One wave32 per batch row; lane n = capsule n.
// Each lane holds its 16-dim capsule vector in registers; all cross-capsule
// reductions are shfl_xor butterflies. 3 routing iterations, write 16 floats.
// ---------------------------------------------------------------------------
__global__ __launch_bounds__(256) void capsule_route(
    const float* __restrict__ hat, float* __restrict__ out) {
  const int lane = threadIdx.x & 31;
  const int row  = blockIdx.x * 8 + (threadIdx.x >> 5);

  const float* hp = hat + (size_t)row * N_TOT + lane * DCAP;
  v4f h0 = *(const v4f*)(hp + 0);
  v4f h1 = *(const v4f*)(hp + 4);
  v4f h2 = *(const v4f*)(hp + 8);
  v4f h3 = *(const v4f*)(hp + 12);
  float h[16];
#pragma unroll
  for (int j = 0; j < 4; ++j) {
    h[j] = h0[j]; h[4 + j] = h1[j]; h[8 + j] = h2[j]; h[12 + j] = h3[j];
  }

  float b = 0.0f;
  float o[16];
#pragma unroll
  for (int it = 0; it < 3; ++it) {
    // softmax over the 32 capsules (lanes)
    float m = b;
    for (int off = 16; off; off >>= 1) m = fmaxf(m, __shfl_xor(m, off));
    float e = expf(b - m);
    float sum = e;
    for (int off = 16; off; off >>= 1) sum += __shfl_xor(sum, off);
    float c = e / sum;

    // s[d] = sum_n c_n * hat[n][d]  (butterfly -> every lane has full s)
    float s[16];
#pragma unroll
    for (int d = 0; d < 16; ++d) s[d] = c * h[d];
    for (int off = 16; off; off >>= 1) {
#pragma unroll
      for (int d = 0; d < 16; ++d) s[d] += __shfl_xor(s[d], off);
    }

    // squash
    float s2 = 0.0f;
#pragma unroll
    for (int d = 0; d < 16; ++d) s2 += s[d] * s[d];
    float scale = s2 / (1.0f + s2) / sqrtf(s2 + 1e-7f);
#pragma unroll
    for (int d = 0; d < 16; ++d) o[d] = scale * s[d];

    // agreement update (skip on last iteration; result identical to ref)
    if (it < 2) {
      float agr = 0.0f;
#pragma unroll
      for (int d = 0; d < 16; ++d) agr += h[d] * o[d];
      b += agr;
    }
  }

  if (lane == 0) {
    float* op = out + (size_t)row * DCAP;
    *(v4f*)(op + 0)  = (v4f){o[0],  o[1],  o[2],  o[3]};
    *(v4f*)(op + 4)  = (v4f){o[4],  o[5],  o[6],  o[7]};
    *(v4f*)(op + 8)  = (v4f){o[8],  o[9],  o[10], o[11]};
    *(v4f*)(op + 12) = (v4f){o[12], o[13], o[14], o[15]};
  }
}

// ---------------------------------------------------------------------------
extern "C" void kernel_launch(void* const* d_in, const int* in_sizes, int n_in,
                              void* d_out, int out_size, void* d_ws, size_t ws_size,
                              hipStream_t stream) {
  const float* x    = (const float*)d_in[0];   // [16384, 1024]
  const float* kern = (const float*)d_in[1];   // [1024, 512]
  float*       out  = (float*)d_out;           // [16384, 16]

  char* ws = (char*)d_ws;
  __bf16* Bhi = (__bf16*)(ws);                         // 1 MB
  __bf16* Blo = (__bf16*)(ws + (size_t)(1u << 20));    // 1 MB
  float*  hat = (float*)(ws + (size_t)(2u << 20));     // 32 MB

  // B split+swizzle: 32 kt * 32 nt * 32 lanes threads
  capsule_swizzle_b<<<128, 256, 0, stream>>>(kern, Bhi, Blo);

  // GEMM: grid (N blocks of 256, M blocks of 128)
  dim3 ggrid(N_TOT / 256, M_TOT / 128);
  capsule_gemm<<<ggrid, 256, 0, stream>>>(x, Bhi, Blo, hat);

  // Routing: one wave per row, 8 rows per block
  capsule_route<<<M_TOT / 8, 256, 0, stream>>>(hat, out);
}
